// graphRCNNFeatureExtractor_86517821214096
// MI455X (gfx1250) — matrix operand order repaired and checked by reading, not compiled, not run on hardware
//
#include <hip/hip_runtime.h>
#include <cstdint>

// ---------------------------------------------------------------------------
// NRI-style graph net for MI455X (gfx1250, wave32, WMMA bf16 path).
// B=64, N=64, F=64, K=4, MSG_HID=256, MSG_OUT=64, N_HID=256, E=4032.
// Edges are receiver-major: receiver r owns edges r*63..r*63+62 with senders
// s = j + (j >= r). One workgroup per (b, r) => aggregation is a local column
// sum in LDS, no global atomics.
//
// GEMM loop nest keeps B (weight) fragments register-resident per k-step and
// iterates over M strips, so each block reads each weight matrix ~once from
// L2 (weights are L2-resident: ~0.6 MB total in bf16, L2 = 192 MB).
// ---------------------------------------------------------------------------

typedef __bf16 bf16;
typedef __attribute__((ext_vector_type(16))) __bf16 v16bf;
typedef __attribute__((ext_vector_type(8)))  float  v8f;

union Frag {
  v16bf v;
  uint4 q[2];
};

static __device__ __forceinline__ bf16 f2bf(float f) {
  union { float f; uint32_t u; } c; c.f = f;
  uint32_t r = (c.u + 0x7FFFu + ((c.u >> 16) & 1u)) >> 16;  // round-nearest-even
  union { unsigned short s; bf16 b; } o; o.s = (unsigned short)r;
  return o.b;
}

// A fragment (16x32 bf16, M x K) from row-major [rows][ld] source at (m0,k0).
// ISA layout: lanes 0-15 = rows, halves 0..7 = K k0..k0+7, halves 8..15 =
// K k0+16..k0+23; lanes 16-31 shift K by +8.
static __device__ __forceinline__ Frag load_A(const bf16* A, int m0, int k0,
                                              int ld, int lane) {
  int row = m0 + (lane & 15);
  int kk  = k0 + ((lane >> 4) << 3);
  const bf16* p = A + (size_t)row * ld + kk;
  Frag f;
  f.q[0] = *(const uint4*)(p);       // K = kk .. kk+7
  f.q[1] = *(const uint4*)(p + 16);  // K = kk+16 .. kk+23
  return f;
}

// B fragment (32x16 bf16, K x N). Weights stored row-major [N][Kdim]
// (reference layout W[out][in]) == column-major B => contiguous K per lane.
// Lanes 0-15: col n, K k0..k0+7, k0+16..k0+23; lanes 16-31 shift K by +8.
static __device__ __forceinline__ Frag load_B(const bf16* W, int n0, int k0,
                                              int ld, int lane) {
  int col = n0 + (lane & 15);
  int kk  = k0 + ((lane >> 4) << 3);
  const bf16* p = W + (size_t)col * ld + kk;
  Frag f;
  f.q[0] = *(const uint4*)(p);      // K = kk .. kk+7
  f.q[1] = *(const uint4*)(p + 16);  // K = kk+16 .. kk+23
  return f;
}

// Wave computes an MT x NT grid of 16x16 tiles. B fragments are loaded once
// per k-step and reused across all MT strips (weights read ~once per block).
template <int KD, int MT, int NT>
static __device__ __forceinline__ void gemm_block(
    const bf16* __restrict__ Alds, int m0base, const bf16* __restrict__ Bg,
    int n0base, v8f* acc, int lane) {
#pragma unroll
  for (int ks = 0; ks < KD / 32; ++ks) {
    Frag bfr[NT];
#pragma unroll
    for (int t = 0; t < NT; ++t)
      bfr[t] = load_B(Bg, n0base + t * 16, ks * 32, KD, lane);
#pragma unroll
    for (int m = 0; m < MT; ++m) {
      Frag a = load_A(Alds, m0base + m * 16, ks * 32, KD, lane);
#pragma unroll
      for (int t = 0; t < NT; ++t)
        acc[m * NT + t] = __builtin_amdgcn_wmma_f32_16x16x32_bf16(
            false, a.v, false, bfr[t].v, (short)0, acc[m * NT + t], false,
            false);
    }
  }
}

// C/D f32 tile layout: lane l, VGPR j -> (row m0 + j + 8*(l>=16), col n0 + l%16)
static __device__ __forceinline__ void store_relu_bf16(
    bf16* out, int ld, int m0, int n0, const v8f& acc,
    const float* __restrict__ bias, int lane) {
  int col  = n0 + (lane & 15);
  int rowb = m0 + ((lane >> 4) << 3);
  float bv = bias[col];
#pragma unroll
  for (int j = 0; j < 8; ++j) {
    float v = acc[j] + bv;
    v = v > 0.f ? v : 0.f;
    out[(size_t)(rowb + j) * ld + col] = f2bf(v);
  }
}

// ---------------------------------------------------------------------------
// Kernel 1: fp32 -> bf16 conversions.
// ---------------------------------------------------------------------------
__global__ __launch_bounds__(256) void convert_kernel(
    const float* __restrict__ inputs, const float* __restrict__ W1,
    const float* __restrict__ W2, const float* __restrict__ Wo1,
    const float* __restrict__ Wo2, const float* __restrict__ Wo3,
    bf16* __restrict__ aug, bf16* __restrict__ w1b, bf16* __restrict__ w2b,
    bf16* __restrict__ wo1b, bf16* __restrict__ wo2b, bf16* __restrict__ wo3b) {
  int stride = gridDim.x * blockDim.x;
  int g = blockIdx.x * blockDim.x + threadIdx.x;
  for (int i = g; i < 4096 * 64; i += stride)
    aug[(size_t)(i >> 6) * 128 + (i & 63)] = f2bf(inputs[i]);
  for (int i = g; i < 4 * 256 * 128; i += stride) w1b[i] = f2bf(W1[i]);
  for (int i = g; i < 4 * 64 * 256;  i += stride) w2b[i] = f2bf(W2[i]);
  for (int i = g; i < 256 * 128;     i += stride) wo1b[i] = f2bf(Wo1[i]);
  for (int i = g; i < 256 * 256;     i += stride) wo2b[i] = f2bf(Wo2[i]);
  for (int i = g; i < 64 * 256;      i += stride) wo3b[i] = f2bf(Wo3[i]);
}

// ---------------------------------------------------------------------------
// Kernel 2: edge MLP + aggregation. One block per (b, r). 256 threads = 8 waves.
// ---------------------------------------------------------------------------
__global__ __launch_bounds__(256) void edge_kernel(
    bf16* __restrict__ aug,              // [4096][128]: read cols 0..63, write 64..127
    const float* __restrict__ rel_type,  // [B][E][4]
    const bf16* __restrict__ W1b,        // [4][256][128]
    const float* __restrict__ b1,        // [4][256]
    const bf16* __restrict__ W2b,        // [4][64][256]
    const float* __restrict__ b2) {      // [4][64]
  __shared__ __align__(16) bf16 preA[64 * 128];  // pre_msg tile (A, K=128)
  __shared__ __align__(16) bf16 h1[64 * 256];    // hidden tile  (A, K=256)
  __shared__ float relL[64 * 4];
  __shared__ float aggL[64];

  const int tid  = threadIdx.x;
  const int lane = tid & 31;
  const int w    = tid >> 5;
  const int bi   = blockIdx.x;
  const int b    = bi >> 6;
  const int r    = bi & 63;

  // ---- stage pre_msg tile: row j<63 -> edge r*63+j (recv=r, send=j+(j>=r))
  {
    const unsigned* aw = (const unsigned*)aug;  // 64 words per aug row
    unsigned* pw = (unsigned*)preA;             // 64 words per preA row
    const int rbase = (b * 64 + r) * 64;
    for (int i = tid; i < 64 * 32; i += 256) {  // receiver half (cols 0..63)
      int row = i >> 5, c = i & 31;
      pw[row * 64 + c] = (row < 63) ? aw[rbase + c] : 0u;
    }
    for (int i = tid; i < 64 * 32; i += 256) {  // sender half (cols 64..127)
      int row = i >> 5, c = i & 31;
      unsigned v = 0u;
      if (row < 63) {
        int s = row + (row >= r);
        v = aw[(b * 64 + s) * 64 + c];
      }
      pw[row * 64 + 32 + c] = v;
    }
    // rel_type tile [row][k] (padded row -> 0 kills the padded row entirely)
    {
      int row = tid >> 2, k = tid & 3;
      relL[tid] = (row < 63)
          ? rel_type[((size_t)b * 4032 + r * 63 + row) * 4 + k] : 0.f;
    }
    if (tid < 64) aggL[tid] = 0.f;
  }
  __syncthreads();

  // GEMM1: wave owns 2 N-tiles (cols w*32..w*32+31), all 4 M strips.
  const int n1 = w * 32;
  // GEMM2: wave owns 1 N-tile (col (w>>1)*16), 2 M strips from (w&1)*32.
  const int m2 = (w & 1) * 32;
  const int n2 = (w >> 1) * 16;

  const v8f vzero = {0.f, 0.f, 0.f, 0.f, 0.f, 0.f, 0.f, 0.f};
  v8f msg[2];
  msg[0] = vzero; msg[1] = vzero;

  for (int k = 0; k < 4; ++k) {
    // ---- GEMM1: h1 = relu(pre_msg * W1[k]^T + b1[k])  (64x128x256)
    v8f acc1[8];
#pragma unroll
    for (int t = 0; t < 8; ++t) acc1[t] = vzero;
    gemm_block<128, 4, 2>(preA, 0, W1b + (size_t)k * 256 * 128, n1, acc1, lane);
#pragma unroll
    for (int m = 0; m < 4; ++m)
#pragma unroll
      for (int t = 0; t < 2; ++t)
        store_relu_bf16(h1, 256, m * 16, n1 + t * 16, acc1[m * 2 + t],
                        b1 + k * 256, lane);
    __syncthreads();

    // ---- GEMM2: h2 = relu(h1 * W2[k]^T + b2[k])  (64x256x64), rel_type gate
    v8f acc2[2];
    acc2[0] = vzero; acc2[1] = vzero;
    gemm_block<256, 2, 1>(h1, m2, W2b + (size_t)k * 64 * 256, n2, acc2, lane);

    const float bv = b2[k * 64 + n2 + (lane & 15)];
#pragma unroll
    for (int i = 0; i < 2; ++i) {
      const int rowb = m2 + i * 16 + ((lane >> 4) << 3);
#pragma unroll
      for (int j = 0; j < 8; ++j) {
        float v = acc2[i][j] + bv;
        v = v > 0.f ? v : 0.f;
        msg[i][j] += relL[(rowb + j) * 4 + k] * v;
      }
    }
    __syncthreads();  // h1 consumed; next edge type may overwrite it
  }

  // ---- aggregate: agg[b,r,c] = column sum of msg over the 64 edge rows
  {
    float p = 0.f;
#pragma unroll
    for (int i = 0; i < 2; ++i)
#pragma unroll
      for (int j = 0; j < 8; ++j) p += msg[i][j];
    atomicAdd(&aggL[n2 + (lane & 15)], p);  // ds_add_f32
  }
  __syncthreads();

  if (tid < 64)
    aug[(size_t)(b * 64 + r) * 128 + 64 + tid] = f2bf(aggL[tid]);
}

// ---------------------------------------------------------------------------
// Kernel 3: node MLP. 64 blocks x 64 node-rows. out = inputs + MLP(aug).
// aug tile staged with GLOBAL_LOAD_ASYNC_TO_LDS_B128 (ASYNCcnt path).
// ---------------------------------------------------------------------------
__global__ __launch_bounds__(256) void node_kernel(
    const bf16* __restrict__ aug,   // [4096][128]
    const float* __restrict__ inputs,
    const bf16* __restrict__ Wo1b, const float* __restrict__ bo1,
    const bf16* __restrict__ Wo2b, const float* __restrict__ bo2,
    const bf16* __restrict__ Wo3b, const float* __restrict__ bo3,
    float* __restrict__ out) {
  __shared__ __align__(16) bf16 augL[64 * 128];
  __shared__ __align__(16) bf16 hL[64 * 256];

  const int tid  = threadIdx.x;
  const int lane = tid & 31;
  const int w    = tid >> 5;
  const int rows0 = blockIdx.x * 64;

  // ---- async global->LDS verbatim copy of the 16KB aug tile
  {
    const char* gbase = (const char*)(aug + (size_t)rows0 * 128);
    unsigned lbase = (unsigned)(uintptr_t)(void*)augL;  // LDS addr = low 32 bits
#pragma unroll
    for (int it = 0; it < 4; ++it) {
      int i = tid + it * 256;  // 1024 chunks of 16B
      unsigned long long ga = (unsigned long long)(uintptr_t)(gbase + i * 16);
      unsigned la = lbase + i * 16;
      asm volatile("global_load_async_to_lds_b128 %0, %1, off"
                   :: "v"(la), "v"(ga) : "memory");
    }
    asm volatile("s_wait_asynccnt 0" ::: "memory");
  }
  __syncthreads();

  const int n1 = w * 32;          // layers 1&2: wave owns 2 N-tiles, 4 M strips
  const int m3 = (w & 1) * 32;    // layer 3: 2 M strips
  const int n3 = (w >> 1) * 16;   // layer 3: 1 N-tile
  const v8f vzero = {0.f, 0.f, 0.f, 0.f, 0.f, 0.f, 0.f, 0.f};

  // layer 1: 64x128x256
  v8f a1[8];
#pragma unroll
  for (int t = 0; t < 8; ++t) a1[t] = vzero;
  gemm_block<128, 4, 2>(augL, 0, Wo1b, n1, a1, lane);
#pragma unroll
  for (int m = 0; m < 4; ++m)
#pragma unroll
    for (int t = 0; t < 2; ++t)
      store_relu_bf16(hL, 256, m * 16, n1 + t * 16, a1[m * 2 + t], bo1, lane);
  __syncthreads();

  // layer 2: 64x256x256 (read hL fully, then overwrite in place)
  v8f a2[8];
#pragma unroll
  for (int t = 0; t < 8; ++t) a2[t] = vzero;
  gemm_block<256, 4, 2>(hL, 0, Wo2b, n1, a2, lane);
  __syncthreads();
#pragma unroll
  for (int m = 0; m < 4; ++m)
#pragma unroll
    for (int t = 0; t < 2; ++t)
      store_relu_bf16(hL, 256, m * 16, n1 + t * 16, a2[m * 2 + t], bo2, lane);
  __syncthreads();

  // layer 3: 64x256x64 + residual
  v8f a3[2];
  a3[0] = vzero; a3[1] = vzero;
  gemm_block<256, 2, 1>(hL, m3, Wo3b, n3, a3, lane);

  const int c = n3 + (lane & 15);
  const float bv = bo3[c];
#pragma unroll
  for (int i = 0; i < 2; ++i) {
    const int rowb = m3 + i * 16 + ((lane >> 4) << 3);
#pragma unroll
    for (int j = 0; j < 8; ++j) {
      size_t gi = (size_t)(rows0 + rowb + j) * 64 + c;
      out[gi] = inputs[gi] + a3[i][j] + bv;
    }
  }
}

// ---------------------------------------------------------------------------
extern "C" void kernel_launch(void* const* d_in, const int* in_sizes, int n_in,
                              void* d_out, int out_size, void* d_ws,
                              size_t ws_size, hipStream_t stream) {
  const float* inputs   = (const float*)d_in[0];
  const float* rel_type = (const float*)d_in[1];
  // d_in[2]=rel_rec, d_in[3]=rel_send: structure exploited analytically.
  const float* W1  = (const float*)d_in[4];
  const float* b1  = (const float*)d_in[5];
  const float* W2  = (const float*)d_in[6];
  const float* b2  = (const float*)d_in[7];
  const float* Wo1 = (const float*)d_in[8];
  const float* bo1 = (const float*)d_in[9];
  const float* Wo2 = (const float*)d_in[10];
  const float* bo2 = (const float*)d_in[11];
  const float* Wo3 = (const float*)d_in[12];
  const float* bo3 = (const float*)d_in[13];
  float* out = (float*)d_out;

  char* ws = (char*)d_ws;
  bf16* aug  = (bf16*)(ws);                    // 4096*128*2 = 1048576 B
  bf16* w1b  = (bf16*)(ws + 1048576);          // 262144 B
  bf16* w2b  = (bf16*)(ws + 1310720);          // 131072 B
  bf16* wo1b = (bf16*)(ws + 1441792);          // 65536 B
  bf16* wo2b = (bf16*)(ws + 1507328);          // 131072 B
  bf16* wo3b = (bf16*)(ws + 1638400);          // 32768 B

  convert_kernel<<<256, 256, 0, stream>>>(inputs, W1, W2, Wo1, Wo2, Wo3, aug,
                                          w1b, w2b, wo1b, wo2b, wo3b);
  edge_kernel<<<4096, 256, 0, stream>>>(aug, rel_type, w1b, b1, w2b, b2);
  node_kernel<<<64, 256, 0, stream>>>(aug, inputs, wo1b, bo1, wo2b, bo2, wo3b,
                                      bo3, out);
}
